// LinearAttention_15221364097788
// MI455X (gfx1250) — matrix operand order repaired and checked by reading, not compiled, not run on hardware
//
#include <hip/hip_runtime.h>

typedef __bf16 bf16_t;
typedef __attribute__((ext_vector_type(16))) __bf16 bf16x16;
typedef __attribute__((ext_vector_type(8)))  __bf16 bf16x8;
typedef __attribute__((ext_vector_type(8)))  float  f32x8;

#define BATCH 32
#define NH 64           // H
#define NW 64           // W
#define CIN 256
#define HID 128
#define M_TOTAL (BATCH * NH * NW)   // 131072

// ---------------- CDNA5 async global->LDS (CDNA5 ISA, 08_async) -----

// Per-lane: copy 16 bytes MEM[g] -> LDS[lds_off]. Tracked by ASYNCcnt.
__device__ __forceinline__ void async_b128(void* lds_dst, const void* g) {
  // Flat LDS aperture: bits [31:0] of the flat address are the LDS offset.
  unsigned lds_off = (unsigned)(uintptr_t)lds_dst;
  asm volatile("global_load_async_to_lds_b128 %0, %1, off"
               :: "v"(lds_off), "v"(g) : "memory");
}
__device__ __forceinline__ void wait_async0() {
  asm volatile("s_wait_asynccnt 0x0" ::: "memory");
}

// ---------------- WMMA helpers (layouts per CDNA5 ISA) -----------

__device__ __forceinline__ f32x8 wmma_bf16(bf16x16 a, bf16x16 b, f32x8 c) {
  return __builtin_amdgcn_wmma_f32_16x16x32_bf16(
      /*neg_a=*/false, a, /*neg_b=*/false, b,
      /*c_mod=*/(short)0, c, /*reuse_a=*/false, /*reuse_b=*/false);
}

// A fragment 16x32 bf16 from LDS tile (row-major, leading dim `ld` elements).
// lane<16: row=lane, K = 0..15; lane>=16: row=lane-16, K = 16..31
__device__ __forceinline__ bf16x16 ldsA(const bf16_t* base, int ld, int lane) {
  const int row = lane & 15, ph = lane >> 4;
  const bf16_t* p = base + row * ld;
  union { bf16x16 v; struct { bf16x8 lo, hi; } s; } u;
  u.s.lo = *(const bf16x8*)(p + ph * 16);
  u.s.hi = *(const bf16x8*)(p + ph * 16 + 8);
  return u.v;
}

// B fragment 32x16 bf16 from LDS stored transposed as [N][K] (leading dim `ld`).
// lane<16: col=lane, K=0..15; lane>=16: col=lane-16, K=16..31  -> one 32B read
__device__ __forceinline__ bf16x16 ldsB(const bf16_t* base, int ld, int lane) {
  const int row = lane & 15, ph = lane >> 4;
  return *(const bf16x16*)(base + row * ld + ph * 16);
}

// ---------------- Kernel W: weight convert + transpose to bf16 [N][K] -------

__global__ __launch_bounds__(256) void k_weights(
    const float* __restrict__ w_qkv, const float* __restrict__ w_out,
    bf16_t* __restrict__ wq_t, bf16_t* __restrict__ wo_t) {
  int i = blockIdx.x * 256 + threadIdx.x;
  if (i < CIN * 384) {                       // w_qkv [256][384] -> wq_t [384][256]
    int k = i / 384, n = i % 384;
    wq_t[n * CIN + k] = (bf16_t)w_qkv[i];
  } else {
    int j = i - CIN * 384;
    if (j < HID * CIN) {                     // w_out [128][256] -> wo_t [256][128]
      int k = j / CIN, n = j % CIN;
      wo_t[n * HID + k] = (bf16_t)w_out[j];
    }
  }
}

// ---------------- Kernel A: qkv = x @ w_qkv  (M=131072, K=256, N=384) -------
// q,k (n<256) -> qk[m][256] bf16 ; v (n>=256) -> vs[b][h][e][n] bf16

__global__ __launch_bounds__(256) void k_qkv_gemm(
    const float* __restrict__ x, const bf16_t* __restrict__ wq_t,
    bf16_t* __restrict__ qk, bf16_t* __restrict__ vs) {
  __shared__ __align__(32) bf16_t a_lds[64 * 32];
  __shared__ __align__(32) bf16_t b_lds[64 * 32];
  const int tid = threadIdx.x, lane = tid & 31, wid = tid >> 5;
  const int wm = wid & 3, wn = wid >> 2;
  const long m0 = (long)blockIdx.x * 64;
  const int  n0 = blockIdx.y * 64;
  const int arow = tid >> 2, acol = (tid & 3) * 8;
  f32x8 acc0 = {}, acc1 = {};

  for (int k0 = 0; k0 < CIN; k0 += 32) {
    // B tile via async DMA (bf16, no conversion needed)
    async_b128(b_lds + arow * 32 + acol, wq_t + (n0 + arow) * CIN + k0 + acol);
    // A tile: x f32 -> bf16 through VGPRs (conversion required)
    const float* xs = x + (m0 + arow) * CIN + k0 + acol;
    float4 f0 = *(const float4*)(xs);
    float4 f1 = *(const float4*)(xs + 4);
    bf16_t* ad = a_lds + arow * 32 + acol;
    ad[0] = (bf16_t)f0.x; ad[1] = (bf16_t)f0.y; ad[2] = (bf16_t)f0.z; ad[3] = (bf16_t)f0.w;
    ad[4] = (bf16_t)f1.x; ad[5] = (bf16_t)f1.y; ad[6] = (bf16_t)f1.z; ad[7] = (bf16_t)f1.w;
    wait_async0();
    __syncthreads();
    bf16x16 a  = ldsA(a_lds + wm * 16 * 32, 32, lane);
    bf16x16 b0 = ldsB(b_lds + (wn * 2 + 0) * 16 * 32, 32, lane);
    bf16x16 b1 = ldsB(b_lds + (wn * 2 + 1) * 16 * 32, 32, lane);
    acc0 = wmma_bf16(a, b0, acc0);
    acc1 = wmma_bf16(a, b1, acc1);
    __syncthreads();
  }

  const int ph = lane >> 4, col = lane & 15;
  if (n0 < 256) {                       // q,k part
#pragma unroll
    for (int r = 0; r < 8; ++r) {
      long m = m0 + wm * 16 + r + 8 * ph;
      int  n = n0 + wn * 32 + col;
      qk[m * 256 + n]      = (bf16_t)acc0[r];
      qk[m * 256 + n + 16] = (bf16_t)acc1[r];
    }
  } else {                              // v part -> vs[b][h][e][nw]
#pragma unroll
    for (int r = 0; r < 8; ++r) {
      long m  = m0 + wm * 16 + r + 8 * ph;
      int  b  = (int)(m >> 12);
      int  md = (int)(m & 4095);
      int  e  = md >> 6, nw = md & 63;
      int  h0 = n0 - 256 + wn * 32 + col;
      vs[(((long)b * HID + h0)      * 64 + e) * 64 + nw] = (bf16_t)acc0[r];
      vs[(((long)b * HID + h0 + 16) * 64 + e) * 64 + nw] = (bf16_t)acc1[r];
    }
  }
}

// ---------------- Kernel B: softmaxes + transposes --------------------------
// block = (d, b), 128 threads. q: softmax over n (W), write Q^T [b][h][n][d].
// k: softmax over h, write [b][h][d][n].

__global__ __launch_bounds__(128) void k_softmax(
    const bf16_t* __restrict__ qk, bf16_t* __restrict__ qs, bf16_t* __restrict__ ks) {
  __shared__ bf16_t t[64 * 256];   // 32 KB: 64 n-rows x 256 channels
  const int d = blockIdx.x, b = blockIdx.y, tid = threadIdx.x;
  const bf16_t* src = qk + ((long)b * 4096 + (long)d * 64) * 256;
  for (int i = tid; i < 2048; i += 128)
    async_b128(t + i * 8, src + i * 8);
  wait_async0();
  __syncthreads();

  // q path: thread = channel h (0..127); values t[n*256 + h], n = 0..63
  {
    float mx = -1e30f;
#pragma unroll 4
    for (int n = 0; n < 64; ++n) mx = fmaxf(mx, (float)t[n * 256 + tid]);
    float s = 0.f;
#pragma unroll 4
    for (int n = 0; n < 64; ++n) s += __expf((float)t[n * 256 + tid] - mx);
    const float inv = 0.17677669529663689f / s;   // scale = 32^-0.5
    bf16_t* qd = qs + ((long)b * HID + tid) * 4096 + d;   // [b][h][n][d]
#pragma unroll 4
    for (int n = 0; n < 64; ++n)
      qd[n * 64] = (bf16_t)(__expf((float)t[n * 256 + tid] - mx) * inv);
  }

  // k path: thread = n (0..63); values t[n*256 + 128 + h], h = 0..127
  if (tid < 64) {
    const bf16_t* kr = t + tid * 256 + 128;
    float mx = -1e30f;
#pragma unroll 4
    for (int h = 0; h < 128; ++h) mx = fmaxf(mx, (float)kr[h]);
    float s = 0.f;
#pragma unroll 4
    for (int h = 0; h < 128; ++h) s += __expf((float)kr[h] - mx);
    const float inv = 1.f / s;
#pragma unroll 4
    for (int h = 0; h < 128; ++h)
      ks[((long)b * HID + h) * 4096 + d * 64 + tid] =
          (bf16_t)(__expf((float)kr[h] - mx) * inv);   // [b][h][d][n]
  }
}

// ---------------- Kernel C: batched attention GEMMs per (b,h) ---------------
// context[d][e] = sum_n K[d][n] V[e][n] ; out[e][n] = sum_d context[d][e] Q^T[n][d]

__global__ __launch_bounds__(256) void k_attn(
    const bf16_t* __restrict__ qs, const bf16_t* __restrict__ ks,
    const bf16_t* __restrict__ vs, bf16_t* __restrict__ attn) {
  __shared__ __align__(32) bf16_t K_s[4096], V_s[4096], Q_s[4096], C_s[4096];
  const int h = blockIdx.x, b = blockIdx.y;
  const long base = ((long)b * HID + h) * 4096;
  const int tid = threadIdx.x, lane = tid & 31, wid = tid >> 5;
  const int wm = wid & 3, wn = wid >> 2;

  for (int i = tid; i < 512; i += 256) {     // 3 x 8KB via async DMA
    async_b128(K_s + i * 8, ks + base + i * 8);
    async_b128(V_s + i * 8, vs + base + i * 8);
    async_b128(Q_s + i * 8, qs + base + i * 8);
  }
  wait_async0();
  __syncthreads();

  // GEMM1: A = K_s [d][n], B = V_s [e][n] (already [N][K] layout)
  f32x8 c0 = {}, c1 = {};
#pragma unroll
  for (int kk = 0; kk < 2; ++kk) {
    bf16x16 a  = ldsA(K_s + wm * 16 * 64 + kk * 32, 64, lane);
    bf16x16 b0 = ldsB(V_s + (wn * 2 + 0) * 16 * 64 + kk * 32, 64, lane);
    bf16x16 b1 = ldsB(V_s + (wn * 2 + 1) * 16 * 64 + kk * 32, 64, lane);
    c0 = wmma_bf16(a, b0, c0);
    c1 = wmma_bf16(a, b1, c1);
  }
  const int ph = lane >> 4, col = lane & 15;
#pragma unroll
  for (int r = 0; r < 8; ++r) {         // store transposed: C_s[e][d]
    int drow = wm * 16 + r + 8 * ph;
    int e0 = wn * 32 + col;
    C_s[e0 * 64 + drow]        = (bf16_t)c0[r];
    C_s[(e0 + 16) * 64 + drow] = (bf16_t)c1[r];
  }
  __syncthreads();

  // GEMM2: A = C_s [e][d], B = Q_s [n][d] ([N][K] layout)
  f32x8 o0 = {}, o1 = {};
#pragma unroll
  for (int kk = 0; kk < 2; ++kk) {
    bf16x16 a  = ldsA(C_s + wm * 16 * 64 + kk * 32, 64, lane);
    bf16x16 b0 = ldsB(Q_s + (wn * 2 + 0) * 16 * 64 + kk * 32, 64, lane);
    bf16x16 b1 = ldsB(Q_s + (wn * 2 + 1) * 16 * 64 + kk * 32, 64, lane);
    o0 = wmma_bf16(a, b0, o0);
    o1 = wmma_bf16(a, b1, o1);
  }
#pragma unroll
  for (int r = 0; r < 8; ++r) {         // attn[b][e][n][h]
    int e = wm * 16 + r + 8 * ph;
    int n = wn * 32 + col;
    attn[(((long)b * 64 + e) * 64 + n)      * HID + h] = (bf16_t)o0[r];
    attn[(((long)b * 64 + e) * 64 + n + 16) * HID + h] = (bf16_t)o1[r];
  }
}

// ---------------- Kernel D: y = attn @ w_out + b_out  (K=128, N=256) --------

__global__ __launch_bounds__(256) void k_out_gemm(
    const bf16_t* __restrict__ attn, const bf16_t* __restrict__ wo_t,
    const float* __restrict__ b_out, float* __restrict__ y) {
  __shared__ __align__(32) bf16_t a_lds[64 * 32];
  __shared__ __align__(32) bf16_t b_lds[64 * 32];
  const int tid = threadIdx.x, lane = tid & 31, wid = tid >> 5;
  const int wm = wid & 3, wn = wid >> 2;
  const long m0 = (long)blockIdx.x * 64;
  const int  n0 = blockIdx.y * 64;
  const int arow = tid >> 2, acol = (tid & 3) * 8;
  f32x8 acc0 = {}, acc1 = {};

  for (int k0 = 0; k0 < HID; k0 += 32) {
    // both tiles are bf16 -> pure async DMA, no VGPR round trip
    async_b128(a_lds + arow * 32 + acol, attn + (m0 + arow) * HID + k0 + acol);
    async_b128(b_lds + arow * 32 + acol, wo_t + (n0 + arow) * HID + k0 + acol);
    wait_async0();
    __syncthreads();
    bf16x16 a  = ldsA(a_lds + wm * 16 * 32, 32, lane);
    bf16x16 b0 = ldsB(b_lds + (wn * 2 + 0) * 16 * 32, 32, lane);
    bf16x16 b1 = ldsB(b_lds + (wn * 2 + 1) * 16 * 32, 32, lane);
    acc0 = wmma_bf16(a, b0, acc0);
    acc1 = wmma_bf16(a, b1, acc1);
    __syncthreads();
  }
  const int ph = lane >> 4, col = lane & 15;
#pragma unroll
  for (int r = 0; r < 8; ++r) {
    long m = m0 + wm * 16 + r + 8 * ph;
    int  n = n0 + wn * 32 + col;
    y[m * CIN + n]      = acc0[r] + b_out[n];
    y[m * CIN + n + 16] = acc1[r] + b_out[n + 16];
  }
}

// ---------------- Kernel E: per-batch mean / rstd (deterministic) -----------

__global__ __launch_bounds__(256) void k_stats(const float* __restrict__ y,
                                               float* __restrict__ stats) {
  __shared__ double sb[256], qb[256];
  const int b = blockIdx.x, tid = threadIdx.x;
  const float* p = y + (long)b * 1048576;
  double s = 0.0, q = 0.0;
  for (int i = tid; i < 1048576; i += 256) {
    float v = p[i];
    s += (double)v;
    q += (double)v * (double)v;
  }
  sb[tid] = s; qb[tid] = q;
  __syncthreads();
  for (int st = 128; st > 0; st >>= 1) {
    if (tid < st) { sb[tid] += sb[tid + st]; qb[tid] += qb[tid + st]; }
    __syncthreads();
  }
  if (tid == 0) {
    double mean = sb[0] / 1048576.0;
    double var  = qb[0] / 1048576.0 - mean * mean;
    stats[b * 2]     = (float)mean;
    stats[b * 2 + 1] = (float)(1.0 / sqrt(var + 1e-6));
  }
}

// ---------------- Kernel F: in-place GroupNorm affine -----------------------

__global__ __launch_bounds__(256) void k_norm(
    float* __restrict__ y, const float* __restrict__ stats,
    const float* __restrict__ gsc, const float* __restrict__ gbi) {
  long i = ((long)blockIdx.x * 256 + threadIdx.x) * 4;
  int b = (int)(i >> 20);                 // 1048576 elems per batch
  int c = (int)(i & 255);
  float mean = stats[b * 2], rstd = stats[b * 2 + 1];
  float4 v = *(float4*)(y + i);
  v.x = (v.x - mean) * rstd * gsc[c]     + gbi[c];
  v.y = (v.y - mean) * rstd * gsc[c + 1] + gbi[c + 1];
  v.z = (v.z - mean) * rstd * gsc[c + 2] + gbi[c + 2];
  v.w = (v.w - mean) * rstd * gsc[c + 3] + gbi[c + 3];
  *(float4*)(y + i) = v;
}

// ---------------- Host launch ----------------------------------------------

extern "C" void kernel_launch(void* const* d_in, const int* in_sizes, int n_in,
                              void* d_out, int out_size, void* d_ws, size_t ws_size,
                              hipStream_t stream) {
  const float* x     = (const float*)d_in[0];
  const float* w_qkv = (const float*)d_in[1];
  const float* w_out = (const float*)d_in[2];
  const float* b_out = (const float*)d_in[3];
  const float* gsc   = (const float*)d_in[4];
  const float* gbi   = (const float*)d_in[5];
  float* y = (float*)d_out;

  char* ws = (char*)d_ws;
  size_t off = 0;
  auto take = [&](size_t bytes) -> void* {
    void* p = ws + off;
    off = (off + bytes + 255) & ~(size_t)255;
    return p;
  };
  bf16_t* qk   = (bf16_t*)take((size_t)M_TOTAL * 256 * sizeof(bf16_t));      // 67 MB
  bf16_t* qs   = (bf16_t*)take((size_t)BATCH * HID * 4096 * sizeof(bf16_t)); // 33.5 MB
  bf16_t* ksb  = (bf16_t*)take((size_t)BATCH * HID * 4096 * sizeof(bf16_t));
  bf16_t* vs   = (bf16_t*)take((size_t)BATCH * HID * 4096 * sizeof(bf16_t));
  bf16_t* wq_t = (bf16_t*)take((size_t)384 * 256 * sizeof(bf16_t));
  bf16_t* wo_t = (bf16_t*)take((size_t)256 * 128 * sizeof(bf16_t));
  float*  stats = (float*)take(64 * sizeof(float));
  bf16_t* attn = qk;   // qk is dead after k_softmax; reuse for attn output

  k_weights<<<512, 256, 0, stream>>>(w_qkv, w_out, wq_t, wo_t);
  k_qkv_gemm<<<dim3(M_TOTAL / 64, 6), 256, 0, stream>>>(x, wq_t, qk, vs);
  k_softmax<<<dim3(64, BATCH), 128, 0, stream>>>(qk, qs, ksb);
  k_attn<<<dim3(HID, BATCH), 256, 0, stream>>>(qs, ksb, vs, attn);
  k_out_gemm<<<dim3(M_TOTAL / 64, 4), 256, 0, stream>>>(attn, wo_t, b_out, y);
  k_stats<<<BATCH, 256, 0, stream>>>(y, stats);
  k_norm<<<(M_TOTAL * CIN / 4) / 256, 256, 0, stream>>>(y, stats, gsc, gbi);
}
